// MyBertSelfAttention_34394098106881
// MI455X (gfx1250) — matrix operand order repaired and checked by reading, not compiled, not run on hardware
//
#include <hip/hip_runtime.h>
#include <hip/hip_bf16.h>

typedef __attribute__((ext_vector_type(16))) _Float16 v16h;
typedef __attribute__((ext_vector_type(8)))  _Float16 v8h;
typedef __attribute__((ext_vector_type(4)))  _Float16 v4h;
typedef __attribute__((ext_vector_type(8)))  float    v8f;

#define BATCH 32
#define SEQ   512
#define DM    768
#define NH    12
#define DH    64

static __device__ __forceinline__ v16h cat8(v8h a, v8h b) {
    return __builtin_shufflevector(a, b, 0,1,2,3,4,5,6,7,8,9,10,11,12,13,14,15);
}

static __device__ __forceinline__ v8f wmma16(v16h a, v16h b, v8f c) {
    return __builtin_amdgcn_wmma_f32_16x16x32_f16(false, a, false, b, (short)0, c, false, false);
}

// ---------------------------------------------------------------------------
// Prep: f32 -> f16 conversion of hidden states
// ---------------------------------------------------------------------------
__global__ void convert_x_kernel(const float* __restrict__ X, _Float16* __restrict__ Xh) {
    size_t i = ((size_t)blockIdx.x * 256 + threadIdx.x) * 4;
    float4 v = *(const float4*)(X + i);
    v4h h;
    h.x = (_Float16)v.x; h.y = (_Float16)v.y; h.z = (_Float16)v.z; h.w = (_Float16)v.w;
    *(v4h*)(Xh + i) = h;
}

// ---------------------------------------------------------------------------
// Prep: convert + transpose weights: Wt[mat][n][k] = (f16) W[mat][k][n]
// ---------------------------------------------------------------------------
__global__ void transpose_w_kernel(const float* __restrict__ Wq,
                                   const float* __restrict__ Wk,
                                   const float* __restrict__ Wv,
                                   _Float16* __restrict__ Wt) {
    int mat = blockIdx.z;
    const float* W = (mat == 0) ? Wq : (mat == 1) ? Wk : Wv;
    int idx = blockIdx.x * 256 + threadIdx.x;      // 0 .. 768*768-1
    int k = idx / DM;
    int n = idx % DM;
    Wt[(size_t)mat * DM * DM + (size_t)n * DM + k] = (_Float16)W[(size_t)k * DM + n];
}

// ---------------------------------------------------------------------------
// Fused QKV projection GEMM:  [16384 x 768] x [768 x 768] per matrix.
// One wave computes a 16x64 output tile via 4 f32 WMMA accumulators.
// Outputs:
//   mat 0: Qh[B,H,S,64]  (scaled by 1/sqrt(64))
//   mat 1: Kh[B,H,S,64]
//   mat 2: Vt[B,H,64,S]  (transposed for PV B-matrix loads)
// ---------------------------------------------------------------------------
__global__ void qkv_gemm_kernel(const _Float16* __restrict__ Xh,
                                const _Float16* __restrict__ Wt,
                                const float* __restrict__ bq,
                                const float* __restrict__ bk,
                                const float* __restrict__ bv,
                                _Float16* __restrict__ Qh,
                                _Float16* __restrict__ Kh,
                                _Float16* __restrict__ Vt) {
    const int wave = threadIdx.x >> 5;
    const int lane = threadIdx.x & 31;
    const int lo = lane & 15;
    const int hi = lane >> 4;

    const int mat = blockIdx.z;
    const int gw = blockIdx.x * 4 + wave;          // 0 .. 12287
    const int mtile = gw % 1024;                   // 16-row tile of M=16384
    const int ngrp  = gw / 1024;                   // 0..11, 64 output cols each
    const int n0 = ngrp * 64;

    const _Float16* W = Wt + (size_t)mat * DM * DM;
    const float* biasp = (mat == 0) ? bq : (mat == 1) ? bk : bv;

    v8f acc[4] = {v8f{0}, v8f{0}, v8f{0}, v8f{0}};

    const _Float16* xrow = Xh + (size_t)(mtile * 16 + lo) * DM;

    for (int k0 = 0; k0 < DM; k0 += 32) {
        // A-matrix 16x32 f16: lane lo = row; halves: K = hi*16 + j (contiguous)
        const _Float16* xp = xrow + k0 + hi * 16;
        v8h a0 = *(const v8h*)xp;
        v8h a1 = *(const v8h*)(xp + 8);
        v16h a = cat8(a0, a1);
#pragma unroll
        for (int nt = 0; nt < 4; ++nt) {
            // B-matrix 32x16 f16: N = lane lo; halves: K = hi*16 + j (contiguous)
            const _Float16* wp = W + (size_t)(n0 + nt * 16 + lo) * DM + k0 + hi * 16;
            v16h b = *(const v16h*)wp;
            acc[nt] = wmma16(a, b, acc[nt]);
        }
    }

#pragma unroll
    for (int nt = 0; nt < 4; ++nt) {
        const int ncol = n0 + nt * 16 + lo;
        const int h = ncol >> 6;
        const int d = ncol & 63;
        const float bias = biasp[ncol];
#pragma unroll
        for (int r = 0; r < 8; ++r) {
            const int m = mtile * 16 + r + hi * 8;
            const int b_ = m >> 9;                 // batch
            const int s_ = m & 511;                // seq
            const float v = acc[nt][r] + bias;
            if (mat == 0) {
                Qh[(((size_t)(b_ * NH + h) * SEQ + s_) * DH) + d] = (_Float16)(v * 0.125f);
            } else if (mat == 1) {
                Kh[(((size_t)(b_ * NH + h) * SEQ + s_) * DH) + d] = (_Float16)v;
            } else {
                Vt[(((size_t)(b_ * NH + h) * DH + d) * SEQ) + s_] = (_Float16)v;
            }
        }
    }
}

// ---------------------------------------------------------------------------
// Power-law attention, flash-style (no SxS materialization).
// One wave owns a 16-row q tile; loop over k in 32-wide chunks:
//   scores 16x32 (4 WMMAs), w = (s + 5)^2, rowsum via shfl_xor reduce,
//   masked w -> LDS (f16) -> reload as A-matrix, PV (4 WMMAs into d=64).
// Final: out = acc / (rowsum + 1e-10).
// ---------------------------------------------------------------------------
__global__ void powerlaw_attn_kernel(const _Float16* __restrict__ Qh,
                                     const _Float16* __restrict__ Kh,
                                     const _Float16* __restrict__ Vt,
                                     const float* __restrict__ mask,
                                     float* __restrict__ out) {
    __shared__ _Float16 scr[4][16 * 32];           // per-wave probs staging tile

    const int wave = threadIdx.x >> 5;
    const int lane = threadIdx.x & 31;
    const int lo = lane & 15;
    const int hi = lane >> 4;

    const int qt = blockIdx.x * 4 + wave;          // 0 .. 12287
    const int bh = qt >> 5;                        // (b*H + h), 32 q-tiles each
    const int qtile = qt & 31;
    const int b_ = bh / NH;
    const int h_ = bh % NH;

    const _Float16* Qp = Qh + (size_t)bh * SEQ * DH;
    const _Float16* Kp = Kh + (size_t)bh * SEQ * DH;
    const _Float16* Vp = Vt + (size_t)bh * DH * SEQ;
    const float* mp = mask + (size_t)b_ * SEQ;
    _Float16* sc = scr[wave];

    // Load Q A-matrices once: two 16x32 fragments covering d = 0..63
    v16h a_q[2];
#pragma unroll
    for (int dstep = 0; dstep < 2; ++dstep) {
        const _Float16* qp = Qp + (size_t)(qtile * 16 + lo) * DH + dstep * 32 + hi * 16;
        a_q[dstep] = cat8(*(const v8h*)qp, *(const v8h*)(qp + 8));
    }

    v8f acc[4] = {v8f{0}, v8f{0}, v8f{0}, v8f{0}};
    float rs[8] = {0.f, 0.f, 0.f, 0.f, 0.f, 0.f, 0.f, 0.f};

    for (int kb = 0; kb < SEQ; kb += 32) {
        // ---- scores: two 16x16 tiles (k cols kb..kb+15, kb+16..kb+31) ----
        v8f s[2] = {v8f{0}, v8f{0}};
#pragma unroll
        for (int kt = 0; kt < 2; ++kt) {
#pragma unroll
            for (int dstep = 0; dstep < 2; ++dstep) {
                const _Float16* kp = Kp + (size_t)(kb + kt * 16 + lo) * DH + dstep * 32 + hi * 16;
                v16h bK = *(const v16h*)kp;
                s[kt] = wmma16(a_q[dstep], bK, s[kt]);
            }
        }

        // ---- power-law weights, rowsum reduce, masked weights to LDS ----
        const float m0 = mp[kb + lo];
        const float m1 = mp[kb + 16 + lo];
#pragma unroll
        for (int r = 0; r < 8; ++r) {
            float w0 = s[0][r] + 5.0f; w0 *= w0;
            float w1 = s[1][r] + 5.0f; w1 *= w1;
            float p = w0 + w1;
            p += __shfl_xor(p, 1, 32);
            p += __shfl_xor(p, 2, 32);
            p += __shfl_xor(p, 4, 32);
            p += __shfl_xor(p, 8, 32);
            rs[r] += p;
            const int M = r + hi * 8;
            sc[M * 32 + lo]      = (_Float16)(w0 * m0);
            sc[M * 32 + 16 + lo] = (_Float16)(w1 * m1);
        }
        __syncthreads();

        // ---- reload probs as A-matrix 16x32 ----
        const _Float16* pp = sc + lo * 32 + hi * 16;
        v16h a_p = cat8(*(const v8h*)pp, *(const v8h*)(pp + 8));

        // ---- PV: 4 WMMAs covering d = 0..63 ----
#pragma unroll
        for (int nt = 0; nt < 4; ++nt) {
            const _Float16* vp = Vp + (size_t)(nt * 16 + lo) * SEQ + kb + hi * 16;
            v16h bV = *(const v16h*)vp;
            acc[nt] = wmma16(a_p, bV, acc[nt]);
        }
        __syncthreads();
    }

    // ---- normalize and store: out[b, s, h*64 + d] ----
#pragma unroll
    for (int r = 0; r < 8; ++r) {
        const int M = r + hi * 8;
        const float inv = 1.0f / (rs[r] + 1e-10f);
        const int srow = qtile * 16 + M;
        float* op = out + ((size_t)(b_ * SEQ + srow) * DM) + h_ * DH;
#pragma unroll
        for (int nt = 0; nt < 4; ++nt) {
            op[nt * 16 + lo] = acc[nt][r] * inv;
        }
    }
}

// ---------------------------------------------------------------------------
extern "C" void kernel_launch(void* const* d_in, const int* in_sizes, int n_in,
                              void* d_out, int out_size, void* d_ws, size_t ws_size,
                              hipStream_t stream) {
    const float* X    = (const float*)d_in[0];
    const float* mask = (const float*)d_in[1];
    const float* Wq   = (const float*)d_in[2];
    const float* bq   = (const float*)d_in[3];
    const float* Wk   = (const float*)d_in[4];
    const float* bk   = (const float*)d_in[5];
    const float* Wv   = (const float*)d_in[6];
    const float* bv   = (const float*)d_in[7];
    float* out = (float*)d_out;

    char* ws = (char*)d_ws;
    size_t off = 0;
    _Float16* Xh = (_Float16*)(ws + off); off += (size_t)BATCH * SEQ * DM * 2;   // 25,165,824
    _Float16* Wt = (_Float16*)(ws + off); off += (size_t)3 * DM * DM * 2;        //  3,538,944
    _Float16* Qh = (_Float16*)(ws + off); off += (size_t)BATCH * NH * SEQ * DH * 2;
    _Float16* Kh = (_Float16*)(ws + off); off += (size_t)BATCH * NH * SEQ * DH * 2;
    _Float16* Vt = (_Float16*)(ws + off); off += (size_t)BATCH * NH * DH * SEQ * 2;

    // X: 32*512*768 = 12,582,912 elems; 4 per thread, 256 threads/block
    convert_x_kernel<<<12288, 256, 0, stream>>>(X, Xh);

    // W: 768*768 elems per matrix, 3 matrices
    transpose_w_kernel<<<dim3(2304, 1, 3), 256, 0, stream>>>(Wq, Wk, Wv, Wt);

    // QKV GEMM: 1024 mtiles * 12 ngroups = 12288 waves / 4 per block; z = matrix
    qkv_gemm_kernel<<<dim3(3072, 1, 3), 128, 0, stream>>>(Xh, Wt, bq, bk, bv, Qh, Kh, Vt);

    // Attention: 384 heads * 32 q-tiles = 12288 waves / 4 per block
    powerlaw_attn_kernel<<<3072, 128, 0, stream>>>(Qh, Kh, Vt, mask, out);
}
